// MoeConv_28905129902149
// MI455X (gfx1250) — matrix operand order repaired and checked by reading, not compiled, not run on hardware
//
#include <hip/hip_runtime.h>
#include <hip/hip_bf16.h>
#include <math.h>

#define B_   32
#define CIN  128
#define COUT 128
#define H_   64
#define W_   64
#define E_   4
#define KTOT (CIN * 9)   // 1152

typedef __attribute__((ext_vector_type(16))) __bf16 v16bf;
typedef __attribute__((ext_vector_type(8)))  float  v8f;

__device__ __forceinline__ unsigned short f2bf(float f) {
  unsigned int u = __builtin_bit_cast(unsigned int, f);
  u += 0x7FFFu + ((u >> 16) & 1u);            // round to nearest even
  return (unsigned short)(u >> 16);
}

// ---------------- 1) global average pool:  pooled[b][c] = mean(x[b,c,:,:]) ---
__global__ void pool_kernel(const float* __restrict__ x, float* __restrict__ pooled) {
  __shared__ float red[256];
  const int bc = blockIdx.x;                   // b*CIN + c
  const float* p = x + (size_t)bc * (H_ * W_);
  float s = 0.f;
  for (int i = threadIdx.x; i < H_ * W_; i += 256) s += p[i];
  red[threadIdx.x] = s;
  __syncthreads();
  for (int off = 128; off > 0; off >>= 1) {
    if ((int)threadIdx.x < off) red[threadIdx.x] += red[threadIdx.x + off];
    __syncthreads();
  }
  if (threadIdx.x == 0) pooled[bc] = red[0] * (1.0f / (H_ * W_));
}

// ---------------- 2) gate: sigmoid linear -> top-2 -> normalized dense w ----
__global__ void gate_kernel(const float* __restrict__ pooled,
                            const float* __restrict__ gate_w,
                            const float* __restrict__ gate_b,
                            const float* __restrict__ gate_bias,
                            float* __restrict__ wdense) {
  const int b = threadIdx.x;
  if (b >= B_) return;
  float score[E_], biased[E_];
  for (int e = 0; e < E_; ++e) {
    float s = gate_b[e];
    for (int c = 0; c < CIN; ++c) s += pooled[b * CIN + c] * gate_w[e * CIN + c];
    score[e]  = 1.0f / (1.0f + __expf(-s));
    biased[e] = score[e] + gate_bias[e];
  }
  int i0 = 0;
  for (int e = 1; e < E_; ++e) if (biased[e] > biased[i0]) i0 = e;   // ties -> lowest idx
  int i1 = (i0 == 0) ? 1 : 0;
  for (int e = 0; e < E_; ++e) if (e != i0 && biased[e] > biased[i1]) i1 = e;
  const float w0 = score[i0], w1 = score[i1];
  const float inv = 1.0f / (w0 + w1);          // ROUTE_SCALE == 1
  for (int e = 0; e < E_; ++e)
    wdense[b * E_ + e] = (e == i0) ? w0 * inv : (e == i1) ? w1 * inv : 0.f;
}

// ---------------- 3) effective bias: beff[b][co] ----------------------------
__global__ void beff_kernel(const float* __restrict__ wdense,
                            const float* __restrict__ expert_b,
                            const float* __restrict__ shared_b,
                            float* __restrict__ beff) {
  const int idx = blockIdx.x * 256 + threadIdx.x;
  if (idx >= B_ * COUT) return;
  const int b = idx >> 7, co = idx & 127;
  float v = shared_b[co];
  for (int e = 0; e < E_; ++e) v += wdense[b * E_ + e] * expert_b[e * COUT + co];
  beff[idx] = v;
}

// ---------------- 4) per-sample effective weight (bf16): ---------------------
// weff[b][co][k] = shared_w[co][k] + sum_e wdense[b][e]*expert_w[e][co][k]
__global__ void mix_kernel(const float* __restrict__ wdense,
                           const float* __restrict__ expert_w,
                           const float* __restrict__ shared_w,
                           unsigned short* __restrict__ weff) {
  const int total = B_ * COUT * KTOT;          // 4,718,592
  const int idx = blockIdx.x * 256 + threadIdx.x;
  if (idx >= total) return;
  const int cok = idx % (COUT * KTOT);
  const int b   = idx / (COUT * KTOT);
  const float w0 = wdense[b * E_ + 0], w1 = wdense[b * E_ + 1];
  const float w2 = wdense[b * E_ + 2], w3 = wdense[b * E_ + 3];
  float v = shared_w[cok];
  v += w0 * expert_w[0 * COUT * KTOT + cok];
  v += w1 * expert_w[1 * COUT * KTOT + cok];
  v += w2 * expert_w[2 * COUT * KTOT + cok];
  v += w3 * expert_w[3 * COUT * KTOT + cok];
  weff[idx] = f2bf(v);
}

// ---------------- 5) main conv: per-sample implicit GEMM with WMMA ----------
// out[b] (COUT x HW) = W_eff[b] (COUT x 1152) * im2col(x[b]) (1152 x HW)
// block: 256 thr (8 waves) tiled 4x2: wave = M 32 x N 64; tile M=128 x N=128.
// Per K-chunk(32): cooperative im2col bf16-pair tile in LDS, B frag = 2x
// ds_load_b128, each B feeds two WMMAs (A0/A1) -> 1 ds_load_b128 per WMMA.
__global__ __launch_bounds__(256)
void moe_conv_kernel(const float* __restrict__ x,
                     const unsigned short* __restrict__ weff,
                     const float* __restrict__ beff,
                     float* __restrict__ out) {
  __shared__ unsigned short s_in[64 * 4 * 66];            // 33,792 B raw tile
  __shared__ __align__(16) unsigned int s_b[2][128 * 20]; // 20,480 B pair bufs

  const int b    = blockIdx.x >> 5;
  const int tile = blockIdx.x & 31;
  const int ht0  = tile * 2;
  const int tid  = threadIdx.x;
  const int lane = tid & 31;
  const int wv   = tid >> 5;
  const int l15  = lane & 15;
  const int half = lane >> 4;
  const int Mbase = (wv >> 1) * 32;   // wave rows: 4 groups of 32 couts
  const int Nbase = (wv & 1) * 64;    // wave cols: 2 groups of 64 pixels

  // stage mapping for im2col pair build: 16 kpairs x 16 pixel-groups
  const int skp   = tid & 15;         // kpair 0..15 handled by this thread
  const int spix0 = (tid >> 4) * 8;   // 8 pixels per thread

  v8f zero = {};
  v8f acc[2][4];
  #pragma unroll
  for (int a = 0; a < 2; ++a)
    #pragma unroll
    for (int s = 0; s < 4; ++s) acc[a][s] = zero;

  const unsigned short* arow0 = weff + (size_t)(b * COUT + Mbase      + l15) * KTOT;
  const unsigned short* arow1 = weff + (size_t)(b * COUT + Mbase + 16 + l15) * KTOT;

  int chunk = 0;
  for (int cih = 0; cih < 2; ++cih) {
    __syncthreads();                            // all prior s_in readers done
    // stage channels [cih*64, cih*64+64), zero-padded halo, f32 -> bf16
    for (int i = tid; i < 64 * 4 * 66; i += 256) {
      const int ci  = i / (4 * 66);
      const int rem = i - ci * (4 * 66);
      const int r   = rem / 66;
      const int c   = rem - r * 66;
      const int gh  = ht0 - 1 + r;
      const int gw  = c - 1;
      float v = 0.f;
      if ((unsigned)gh < (unsigned)H_ && (unsigned)gw < (unsigned)W_)
        v = x[(((size_t)b * CIN + cih * 64 + ci) * H_ + gh) * W_ + gw];
      s_in[i] = f2bf(v);
    }
    __syncthreads();

    for (int kc = cih * 576; kc < cih * 576 + 576; kc += 32, ++chunk) {
      unsigned int* sb = s_b[chunk & 1];

      // ---- cooperative im2col pair build: sb[pix*20 + kpair] = {k, k+1} ----
      {
        const int k = kc + 2 * skp;
        int ci = k / 9;
        int rr = k - ci * 9;
        int kh = rr / 3;
        int kw = rr - kh * 3;
        int cil = ci - cih * 64;
        int kh1 = kh, kw1 = kw + 1, cil1 = cil;
        if (kw1 == 3) { kw1 = 0; if (++kh1 == 3) { kh1 = 0; ++cil1; } }
        const int off0 = (cil  * 4 + kh ) * 66 + kw;
        const int off1 = (cil1 * 4 + kh1) * 66 + kw1;
        #pragma unroll
        for (int q = 0; q < 8; ++q) {
          const int pix = spix0 + q;
          const int pr  = (pix >> 6) * 66 + (pix & 63);
          const unsigned int lo = s_in[off0 + pr];
          const unsigned int hi = s_in[off1 + pr];
          sb[pix * 20 + skp] = lo | (hi << 16);
        }
      }
      __syncthreads();   // double-buffered: safe with single barrier per chunk

      // ---- A fragments (ISA 16-bit A 16x32 lane layout), global b128 ----
      union { unsigned int u[8]; v16bf v; } afr0, afr1;
      {
        const unsigned short* ap0 = arow0 + kc + 8 * half;
        const unsigned short* ap1 = arow1 + kc + 8 * half;
        const uint4 q0 = *(const uint4*)(ap0);
        const uint4 q1 = *(const uint4*)(ap0 + 16);
        const uint4 q2 = *(const uint4*)(ap1);
        const uint4 q3 = *(const uint4*)(ap1 + 16);
        afr0.u[0] = q0.x; afr0.u[1] = q0.y; afr0.u[2] = q0.z; afr0.u[3] = q0.w;
        afr0.u[4] = q1.x; afr0.u[5] = q1.y; afr0.u[6] = q1.z; afr0.u[7] = q1.w;
        afr1.u[0] = q2.x; afr1.u[1] = q2.y; afr1.u[2] = q2.z; afr1.u[3] = q2.w;
        afr1.u[4] = q3.x; afr1.u[5] = q3.y; afr1.u[6] = q3.z; afr1.u[7] = q3.w;
      }

      #pragma unroll
      for (int sub = 0; sub < 4; ++sub) {
        const int p = Nbase + sub * 16 + l15;   // pixel column for this lane
        const unsigned int* bp = &sb[p * 20 + 4 * half];
        const uint4 b0 = *(const uint4*)(bp);      // kpairs +0..3
        const uint4 b1 = *(const uint4*)(bp + 8);  // kpairs +8..11
        union { unsigned int u[8]; v16bf v; } bfr;
        bfr.u[0] = b0.x; bfr.u[1] = b0.y; bfr.u[2] = b0.z; bfr.u[3] = b0.w;
        bfr.u[4] = b1.x; bfr.u[5] = b1.y; bfr.u[6] = b1.z; bfr.u[7] = b1.w;
        acc[0][sub] = __builtin_amdgcn_wmma_f32_16x16x32_bf16(
            false, afr0.v, false, bfr.v, (short)0, acc[0][sub], false, false);
        acc[1][sub] = __builtin_amdgcn_wmma_f32_16x16x32_bf16(
            false, afr1.v, false, bfr.v, (short)0, acc[1][sub], false, false);
      }
    }
  }

  // ---- epilogue: += effective bias, store f32 (C layout: M = v + 8*half) ----
  #pragma unroll
  for (int a = 0; a < 2; ++a) {
    #pragma unroll
    for (int sub = 0; sub < 4; ++sub) {
      const int p  = Nbase + sub * 16 + l15;
      const int ph = p >> 6, pw = p & 63;
      const int h  = ht0 + ph;
      #pragma unroll
      for (int v = 0; v < 8; ++v) {
        const int m = Mbase + a * 16 + v + 8 * half;
        const float bv = beff[b * COUT + m];
        out[(((size_t)b * COUT + m) * H_ + h) * W_ + pw] = acc[a][sub][v] + bv;
      }
    }
  }
}

extern "C" void kernel_launch(void* const* d_in, const int* in_sizes, int n_in,
                              void* d_out, int out_size, void* d_ws, size_t ws_size,
                              hipStream_t stream) {
  const float* x         = (const float*)d_in[0];
  const float* gate_w    = (const float*)d_in[1];
  const float* gate_b    = (const float*)d_in[2];
  const float* gate_bias = (const float*)d_in[3];
  const float* expert_w  = (const float*)d_in[4];
  const float* expert_b  = (const float*)d_in[5];
  const float* shared_w  = (const float*)d_in[6];
  const float* shared_b  = (const float*)d_in[7];
  float* out = (float*)d_out;

  char* ws = (char*)d_ws;
  float*          pooled = (float*)(ws);                  // 16384 B
  float*          wdense = (float*)(ws + 16384);          //   512 B
  float*          beff   = (float*)(ws + 16896);          // 16384 B
  unsigned short* weff   = (unsigned short*)(ws + 33280); // 9,437,184 B (bf16)

  pool_kernel<<<B_ * CIN, 256, 0, stream>>>(x, pooled);
  gate_kernel<<<1, 32, 0, stream>>>(pooled, gate_w, gate_b, gate_bias, wdense);
  beff_kernel<<<(B_ * COUT + 255) / 256, 256, 0, stream>>>(wdense, expert_b, shared_b, beff);
  const int total = B_ * COUT * KTOT;
  mix_kernel<<<(total + 255) / 256, 256, 0, stream>>>(wdense, expert_w, shared_w, weff);
  moe_conv_kernel<<<B_ * 32, 256, 0, stream>>>(x, weff, beff, out);
}
